// News_encoder_co1_74612171866751
// MI455X (gfx1250) — matrix operand order repaired and checked by reading, not compiled, not run on hardware
//
#include <hip/hip_runtime.h>
#include <stdint.h>

// ---------------------------------------------------------------------------
// CDNA5 (gfx1250) bf16 WMMA types
// ---------------------------------------------------------------------------
typedef __attribute__((ext_vector_type(16))) __bf16 v16bf;
typedef __attribute__((ext_vector_type(8)))  float  v8f;
typedef __attribute__((ext_vector_type(4)))  int    i32x4;

union FragBF { uint16_t u[16]; uint4 q[2]; v16bf v; };

#define GLOBAL_AS __attribute__((address_space(1)))
#define LDS_AS    __attribute__((address_space(3)))

#if defined(__has_builtin)
#if __has_builtin(__builtin_amdgcn_global_load_async_to_lds_b128) && \
    __has_builtin(__builtin_amdgcn_s_wait_asynccnt)
#define USE_ASYNC_LDS 1
#endif
#endif
#ifndef USE_ASYNC_LDS
#define USE_ASYNC_LDS 0
#endif

__device__ __forceinline__ uint16_t f32_to_bf16(float f) {
  uint32_t u = __float_as_uint(f);
  uint32_t r = u + 0x7FFFu + ((u >> 16) & 1u);   // round to nearest even
  return (uint16_t)(r >> 16);
}
__device__ __forceinline__ float bf2f(uint16_t h) {
  return __uint_as_float(((uint32_t)h) << 16);
}

// ---------------------------------------------------------------------------
// Generic bf16 GEMM:  C[M,N] = A[M,K] * B[K,N] (+bias) (opt tanh)
// A row-major (lda), B row-major (ldb). Requires K%64==0, M%64==0,
// lda/ldb multiples of 8. Block = 256 threads (8 waves), tile 64x64,
// K-stage 64: each wave issues 4 v_wmma per barrier pair (A frag reused).
// Columns >= N are computed on garbage and discarded at the store (epilogue
// guard); all K-pad rows of B are zero so pad activations cannot leak in.
// A tiles are staged via GLOBAL_LOAD_ASYNC_TO_LDS_B128 (ASYNCcnt DMA path);
// B tiles go through VGPRs because they are transposed in flight.
// ---------------------------------------------------------------------------
template<bool OUT_F32, bool ACT_TANH>
__global__ __launch_bounds__(256)
void gemm_bf16(const uint16_t* __restrict__ A, int lda,
               const uint16_t* __restrict__ Bw, int ldb,
               const float* __restrict__ bias,
               void* __restrict__ Cptr, int ldc,
               int M, int N, int K)
{
  __shared__ __align__(16) uint16_t As[64 * 64];   // 8 KB, row-major [m][k]
  __shared__ __align__(16) uint16_t Bs[64 * 64];   // 8 KB, transposed [n][k]

  const int tid  = threadIdx.x;
  const int lane = tid & 31;
  const int wave = tid >> 5;
  const int m0 = blockIdx.y * 64;
  const int n0 = blockIdx.x * 64;
  const int wm  = wave & 3;        // 16-row strip 0..3
  const int wn2 = wave >> 2;       // 32-col half 0..1
  const int half = lane >> 4;
  const int lm   = lane & 15;

  // staging map: thread -> row sr (0..63), col base sc (0,16,32,48)
  const int sr = tid >> 2;
  const int sc = (tid & 3) * 16;

  v8f acc0 = {0.f, 0.f, 0.f, 0.f, 0.f, 0.f, 0.f, 0.f};
  v8f acc1 = {0.f, 0.f, 0.f, 0.f, 0.f, 0.f, 0.f, 0.f};

  for (int k0 = 0; k0 < K; k0 += 64) {
    const uint16_t* aSrc = A  + (size_t)(m0 + sr) * lda + (k0 + sc);
    const uint16_t* bSrc = Bw + (size_t)(k0 + sr) * ldb + (n0 + sc);

    // prefetch next K-stage into L2/WGP$ (global_prefetch_b8)
    if (k0 + 64 < K) {
      __builtin_prefetch(aSrc + 64, 0, 3);
      __builtin_prefetch(Bw + (size_t)(k0 + 64 + sr) * ldb + (n0 + sc), 0, 3);
    }

    // ---- stage A (no transpose): async DMA global->LDS when available ----
#if USE_ASYNC_LDS
    __builtin_amdgcn_global_load_async_to_lds_b128(
        (GLOBAL_AS i32x4*)aSrc, (LDS_AS i32x4*)&As[sr * 64 + sc], 0, 0);
    __builtin_amdgcn_global_load_async_to_lds_b128(
        (GLOBAL_AS i32x4*)aSrc, (LDS_AS i32x4*)&As[sr * 64 + sc], 16, 0);
#else
    *(uint4*)&As[sr * 64 + sc]     = *(const uint4*)aSrc;
    *(uint4*)&As[sr * 64 + sc + 8] = *(const uint4*)(aSrc + 8);
#endif

    // ---- stage B transposed through VGPRs: Bs[n][k] ----
    {
      union { uint4 q; uint16_t u[8]; } t0, t1;
      t0.q = *(const uint4*)bSrc;
      t1.q = *(const uint4*)(bSrc + 8);
      #pragma unroll
      for (int i = 0; i < 8; ++i) Bs[(sc + i) * 64 + sr]     = t0.u[i];
      #pragma unroll
      for (int i = 0; i < 8; ++i) Bs[(sc + 8 + i) * 64 + sr] = t1.u[i];
    }

#if USE_ASYNC_LDS
    __builtin_amdgcn_s_wait_asynccnt(0);
#endif
    __syncthreads();

    #pragma unroll
    for (int kt = 0; kt < 2; ++kt) {
      FragBF fa, fb0, fb1;
      // A frag (16x32): lane lm = row; e -> K = 32kt + 16*half + e
      const int ab = (16 * wm + lm) * 64 + 32 * kt;
      fa.q[0] = *(const uint4*)&As[ab + 16 * half];
      fa.q[1] = *(const uint4*)&As[ab + 16 * half + 8];
      // B frags (32x16): lane lm = col; e -> K = 32kt + 16*half + e
      const int nA = (32 * wn2 + lm) * 64      + 32 * kt + 16 * half;
      const int nB = (32 * wn2 + 16 + lm) * 64 + 32 * kt + 16 * half;
      fb0.q[0] = *(const uint4*)&Bs[nA];
      fb0.q[1] = *(const uint4*)&Bs[nA + 8];
      fb1.q[0] = *(const uint4*)&Bs[nB];
      fb1.q[1] = *(const uint4*)&Bs[nB + 8];
      acc0 = __builtin_amdgcn_wmma_f32_16x16x32_bf16(false, fa.v, false, fb0.v,
                                                     (short)0, acc0, false, false);
      acc1 = __builtin_amdgcn_wmma_f32_16x16x32_bf16(false, fa.v, false, fb1.v,
                                                     (short)0, acc1, false, false);
    }
    __syncthreads();
  }

  // ---- epilogue: C layout vgpr r -> M = mb + r ----
  const int mb = m0 + 16 * wm + 8 * half;
  #pragma unroll
  for (int t = 0; t < 2; ++t) {
    const v8f av = t ? acc1 : acc0;
    const int gn = n0 + 32 * wn2 + 16 * t + lm;
    if (gn < N) {
      const float bv = bias ? bias[gn] : 0.0f;
      #pragma unroll
      for (int r = 0; r < 8; ++r) {
        float v = av[r] + bv;
        if (ACT_TANH) v = tanhf(v);
        const size_t idx = (size_t)(mb + r) * ldc + gn;
        if (OUT_F32) ((float*)Cptr)[idx] = v;
        else         ((uint16_t*)Cptr)[idx] = f32_to_bf16(v);
      }
    }
  }
}

// ---------------------------------------------------------------------------
// Embedding gather + f32 -> bf16 (zero-pads cols >= tdim)
// ---------------------------------------------------------------------------
__global__ void gather_emb(const int* __restrict__ ids,
                           const float* __restrict__ table, int tdim,
                           uint16_t* __restrict__ dst, int ld, int idBase)
{
  const int row = blockIdx.x;
  const int c   = threadIdx.x;                 // blockDim = ld (320)
  const int id  = ids[idBase + row];
  const float v = (c < tdim) ? table[(size_t)id * tdim + c] : 0.0f;
  dst[(size_t)row * ld + c] = f32_to_bf16(v);
}

// Weight convert with K/N zero padding into a (possibly shared) bf16 matrix
__global__ void wconv(const float* __restrict__ src, uint16_t* __restrict__ dst,
                      int K, int Nsrc, int Kpad, int Nreg, int ldb, int colOff)
{
  const int k = blockIdx.x;                    // 0..Kpad-1
  const int n = threadIdx.x;                   // 0..Nreg-1
  const float v = (k < K && n < Nsrc) ? src[(size_t)k * Nsrc + n] : 0.0f;
  dst[(size_t)k * ldb + colOff + n] = f32_to_bf16(v);
}

__global__ void bpad(const float* __restrict__ src, float* __restrict__ dst,
                     int N, int Npad)
{
  const int n = blockIdx.x * blockDim.x + threadIdx.x;
  if (n < Npad) dst[n] = (n < N) ? src[n] : 0.0f;
}

// ---------------------------------------------------------------------------
// Small multi-head attention core (VALU, fp32). One thread per (sample,h,q).
// ---------------------------------------------------------------------------
template<int LQ, int LK, int HH, int DH>
__global__ void attn_kernel(const uint16_t* __restrict__ Qb, int ldq, int qOff,
                            const uint16_t* __restrict__ Kb, int ldk, int kOff,
                            const uint16_t* __restrict__ Vb, int ldv, int vOff,
                            uint16_t* __restrict__ Ob, int ldo, int oOff, int CB)
{
  const int tid = blockIdx.x * blockDim.x + threadIdx.x;
  if (tid >= CB * HH * LQ) return;
  const int s = tid / (HH * LQ);
  const int r = tid % (HH * LQ);
  const int h = r / LQ;
  const int q = r % LQ;

  float qv[DH];
  const uint16_t* qrow = Qb + (size_t)(s * LQ + q) * ldq + qOff + h * DH;
  #pragma unroll
  for (int d = 0; d < DH; ++d) qv[d] = bf2f(qrow[d]);

  const float scale = rsqrtf((float)DH);
  float sc[LK];
  float mx = -1e30f;
  for (int k = 0; k < LK; ++k) {
    const uint16_t* krow = Kb + (size_t)(s * LK + k) * ldk + kOff + h * DH;
    float a = 0.f;
    #pragma unroll
    for (int d = 0; d < DH; ++d) a += qv[d] * bf2f(krow[d]);
    a *= scale;
    sc[k] = a;
    mx = fmaxf(mx, a);
  }
  float sum = 0.f;
  for (int k = 0; k < LK; ++k) { sc[k] = __expf(sc[k] - mx); sum += sc[k]; }
  const float inv = 1.0f / sum;

  float outv[DH];
  #pragma unroll
  for (int d = 0; d < DH; ++d) outv[d] = 0.f;
  for (int k = 0; k < LK; ++k) {
    const float a = sc[k] * inv;
    const uint16_t* vrow = Vb + (size_t)(s * LK + k) * ldv + vOff + h * DH;
    #pragma unroll
    for (int d = 0; d < DH; ++d) outv[d] += a * bf2f(vrow[d]);
  }
  uint16_t* orow = Ob + (size_t)(s * LQ + q) * ldo + oOff + h * DH;
  #pragma unroll
  for (int d = 0; d < DH; ++d) orow[d] = f32_to_bf16(outv[d]);
}

// pool score:  s[row] = tanh-hidden row . w2
__global__ void rowdot(const uint16_t* __restrict__ H, int ldh,
                       const float* __restrict__ w, int N,
                       float* __restrict__ out, int rows)
{
  const int r = blockIdx.x * blockDim.x + threadIdx.x;
  if (r >= rows) return;
  const uint16_t* row = H + (size_t)r * ldh;
  float acc = 0.f;
  for (int i = 0; i < N; ++i) acc += bf2f(row[i]) * w[i];
  out[r] = acc;
}

// softmax over L + weighted sum, writes bf16 into news[b, colOff..colOff+F)
template<int L>
__global__ void poolsum(const float* __restrict__ s,
                        const uint16_t* __restrict__ X, int ldx, int F,
                        uint16_t* __restrict__ news, int ldn, int colOff)
{
  __shared__ float a[L];
  const int b = blockIdx.x;
  const int tid = threadIdx.x;
  if (tid == 0) {
    float tmp[L];
    float mx = -1e30f;
    #pragma unroll
    for (int l = 0; l < L; ++l) { tmp[l] = s[b * L + l]; mx = fmaxf(mx, tmp[l]); }
    float sum = 0.f;
    #pragma unroll
    for (int l = 0; l < L; ++l) { tmp[l] = __expf(tmp[l] - mx); sum += tmp[l]; }
    const float inv = 1.0f / sum;
    #pragma unroll
    for (int l = 0; l < L; ++l) a[l] = tmp[l] * inv;
  }
  __syncthreads();
  for (int f = tid; f < F; f += blockDim.x) {
    float acc = 0.f;
    #pragma unroll
    for (int l = 0; l < L; ++l)
      acc += a[l] * bf2f(X[(size_t)(b * L + l) * ldx + f]);
    news[(size_t)b * ldn + colOff + f] = f32_to_bf16(acc);
  }
}

// news cols 800..999 = vert embedding, 1000..1023 = 0
__global__ void vert_assemble(const int* __restrict__ vert_ids,
                              const float* __restrict__ vt,
                              uint16_t* __restrict__ news, int ldn, int b0)
{
  const int s = blockIdx.x;
  const int c = threadIdx.x;                   // 224 threads
  const int vid = vert_ids[b0 + s];
  const float v = (c < 200) ? vt[(size_t)vid * 200 + c] : 0.0f;
  news[(size_t)s * ldn + 800 + c] = f32_to_bf16(v);
}

// ---------------------------------------------------------------------------
extern "C" void kernel_launch(void* const* d_in, const int* in_sizes, int n_in,
                              void* d_out, int out_size, void* d_ws, size_t ws_size,
                              hipStream_t stream)
{
  (void)in_sizes; (void)n_in; (void)out_size; (void)ws_size;

  const int*   title_ids    = (const int*)  d_in[0];
  const int*   vert_ids     = (const int*)  d_in[1];
  const int*   entity_ids   = (const int*)  d_in[2];
  const float* word_table   = (const float*)d_in[3];
  const float* vert_table   = (const float*)d_in[4];
  const float* entity_table = (const float*)d_in[5];
  const float* WQ   = (const float*)d_in[6];
  const float* WK   = (const float*)d_in[7];
  const float* WV   = (const float*)d_in[8];
  const float* WQ2  = (const float*)d_in[9];
  const float* WK2  = (const float*)d_in[10];
  const float* WV2  = (const float*)d_in[11];
  const float* fc1_W = (const float*)d_in[12];
  const float* fc1_b = (const float*)d_in[13];
  const float* fc2_W = (const float*)d_in[14];
  const float* fc2_b = (const float*)d_in[15];
  const float* apt_W1 = (const float*)d_in[16];
  const float* apt_b1 = (const float*)d_in[17];
  const float* apt_w2 = (const float*)d_in[18];
  const float* ape_W1 = (const float*)d_in[19];
  const float* ape_b1 = (const float*)d_in[20];
  const float* ape_w2 = (const float*)d_in[21];
  const float* fc3_W  = (const float*)d_in[22];
  const float* fc3_b  = (const float*)d_in[23];
  float* out = (float*)d_out;

  enum { Bt = 4096, LT = 30, LE = 10, Dm = 300, Kd = 320,
         CB = 512, NCH = Bt / CB,
         NP = 1824,   // fused projection width: 3*400 + 3*208
         NC = 640,    // concat(self 400, co 200) K-padded to 64
         NV = 448,    // fc1/fc2 out (400) K-padded to 64
         NH = 208,    // pool hidden (200) padded to 16
         NN = 1024,   // news (1000) K-padded
         ON = 400,
         RT = CB * LT, RE = CB * LE };

  // ---- workspace carve-out (chunked so live set ~150 MB < 192 MB L2) ----
  char* ws = (char*)d_ws;
  size_t off = 0;
  auto alloc = [&](size_t bytes) -> char* {
    off = (off + 255) & ~(size_t)255;
    char* p = ws + off;
    off += bytes;
    return p;
  };
  uint16_t* Wallb  = (uint16_t*)alloc((size_t)Kd * NP * 2);
  uint16_t* fc1Wb  = (uint16_t*)alloc((size_t)NC * NV * 2);
  uint16_t* fc2Wb  = (uint16_t*)alloc((size_t)NC * NV * 2);
  uint16_t* aptWb  = (uint16_t*)alloc((size_t)NV * NH * 2);
  uint16_t* apeWb  = (uint16_t*)alloc((size_t)NV * NH * 2);
  uint16_t* fc3Wb  = (uint16_t*)alloc((size_t)NN * ON * 2);
  float*    fc1bp  = (float*)alloc(NV * 4);
  float*    fc2bp  = (float*)alloc(NV * 4);
  float*    aptbp  = (float*)alloc(NH * 4);
  float*    apebp  = (float*)alloc(NH * 4);
  uint16_t* titleE  = (uint16_t*)alloc((size_t)RT * Kd * 2);
  uint16_t* entityE = (uint16_t*)alloc((size_t)RE * Kd * 2);
  uint16_t* projT   = (uint16_t*)alloc((size_t)RT * NP * 2);
  uint16_t* projE   = (uint16_t*)alloc((size_t)RE * NP * 2);
  uint16_t* titleC  = (uint16_t*)alloc((size_t)RT * NC * 2);
  uint16_t* entityC = (uint16_t*)alloc((size_t)RE * NC * 2);
  uint16_t* titleV  = (uint16_t*)alloc((size_t)RT * NV * 2);
  uint16_t* entityV = (uint16_t*)alloc((size_t)RE * NV * 2);
  uint16_t* Ht      = (uint16_t*)alloc((size_t)RT * NH * 2);
  uint16_t* He      = (uint16_t*)alloc((size_t)RE * NH * 2);
  float*    sT      = (float*)alloc((size_t)RT * 4);
  float*    sE      = (float*)alloc((size_t)RE * 4);
  uint16_t* news    = (uint16_t*)alloc((size_t)CB * NN * 2);

  // ---- one-time weight conversion to zero-padded bf16 ----
  wconv<<<Kd, 400, 0, stream>>>(WQ,  Wallb, Dm, 400, Kd, 400, NP, 0);
  wconv<<<Kd, 400, 0, stream>>>(WK,  Wallb, Dm, 400, Kd, 400, NP, 400);
  wconv<<<Kd, 400, 0, stream>>>(WV,  Wallb, Dm, 400, Kd, 400, NP, 800);
  wconv<<<Kd, 208, 0, stream>>>(WQ2, Wallb, Dm, 200, Kd, 208, NP, 1200);
  wconv<<<Kd, 208, 0, stream>>>(WK2, Wallb, Dm, 200, Kd, 208, NP, 1408);
  wconv<<<Kd, 208, 0, stream>>>(WV2, Wallb, Dm, 200, Kd, 208, NP, 1616);
  wconv<<<NC, NV, 0, stream>>>(fc1_W, fc1Wb, 600, 400, NC, NV, NV, 0);
  wconv<<<NC, NV, 0, stream>>>(fc2_W, fc2Wb, 600, 400, NC, NV, NV, 0);
  wconv<<<NV, NH, 0, stream>>>(apt_W1, aptWb, 400, 200, NV, NH, NH, 0);
  wconv<<<NV, NH, 0, stream>>>(ape_W1, apeWb, 400, 200, NV, NH, NH, 0);
  wconv<<<NN, ON, 0, stream>>>(fc3_W, fc3Wb, 1000, 400, NN, ON, ON, 0);
  bpad<<<1, NV, 0, stream>>>(fc1_b, fc1bp, 400, NV);
  bpad<<<1, NV, 0, stream>>>(fc2_b, fc2bp, 400, NV);
  bpad<<<1, NH, 0, stream>>>(apt_b1, aptbp, 200, NH);
  bpad<<<1, NH, 0, stream>>>(ape_b1, apebp, 200, NH);

  for (int c = 0; c < NCH; ++c) {
    const int b0 = c * CB;

    // embeddings -> bf16 (K padded to 320; pad weight rows are zero)
    gather_emb<<<RT, Kd, 0, stream>>>(title_ids,  word_table,   Dm, titleE,  Kd, b0 * LT);
    gather_emb<<<RE, Kd, 0, stream>>>(entity_ids, entity_table, Dm, entityE, Kd, b0 * LE);

    // fused Q|K|V|Q2|K2|V2 projection (one big WMMA GEMM per side)
    gemm_bf16<false, false><<<dim3((NP + 63) / 64, RT / 64), 256, 0, stream>>>(
        titleE,  Kd, Wallb, NP, nullptr, (void*)projT, NP, RT, NP, Kd);
    gemm_bf16<false, false><<<dim3((NP + 63) / 64, RE / 64), 256, 0, stream>>>(
        entityE, Kd, Wallb, NP, nullptr, (void*)projE, NP, RE, NP, Kd);

    // attentions (VALU; tiny fraction of FLOPs)
    attn_kernel<30, 30, 20, 20><<<(CB * 20 * 30 + 255) / 256, 256, 0, stream>>>(
        projT, NP, 0,    projT, NP, 400,  projT, NP, 800,  titleC,  NC, 0,   CB);
    attn_kernel<30, 10, 5, 40><<<(CB * 5 * 30 + 255) / 256, 256, 0, stream>>>(
        projT, NP, 1200, projE, NP, 1408, projE, NP, 1616, titleC,  NC, 400, CB);
    attn_kernel<10, 10, 20, 20><<<(CB * 20 * 10 + 255) / 256, 256, 0, stream>>>(
        projE, NP, 0,    projE, NP, 400,  projE, NP, 800,  entityC, NC, 0,   CB);
    attn_kernel<10, 30, 5, 40><<<(CB * 5 * 10 + 255) / 256, 256, 0, stream>>>(
        projE, NP, 1200, projT, NP, 1408, projT, NP, 1616, entityC, NC, 400, CB);

    // fc1 / fc2 (WMMA)
    gemm_bf16<false, false><<<dim3((NV + 63) / 64, RT / 64), 256, 0, stream>>>(
        titleC,  NC, fc1Wb, NV, fc1bp, (void*)titleV,  NV, RT, NV, NC);
    gemm_bf16<false, false><<<dim3((NV + 63) / 64, RE / 64), 256, 0, stream>>>(
        entityC, NC, fc2Wb, NV, fc2bp, (void*)entityV, NV, RE, NV, NC);

    // pooling hidden = tanh(x @ W1 + b1)  (WMMA + fused tanh)
    gemm_bf16<false, true><<<dim3((NH + 63) / 64, RT / 64), 256, 0, stream>>>(
        titleV,  NV, aptWb, NH, aptbp, (void*)Ht, NH, RT, NH, NV);
    gemm_bf16<false, true><<<dim3((NH + 63) / 64, RE / 64), 256, 0, stream>>>(
        entityV, NV, apeWb, NH, apebp, (void*)He, NH, RE, NH, NV);

    // pooling scores + softmax + weighted sum -> news[:,0:400] / [:,400:800]
    rowdot<<<(RT + 255) / 256, 256, 0, stream>>>(Ht, NH, apt_w2, 200, sT, RT);
    rowdot<<<(RE + 255) / 256, 256, 0, stream>>>(He, NH, ape_w2, 200, sE, RE);
    poolsum<30><<<CB, 128, 0, stream>>>(sT, titleV,  NV, 400, news, NN, 0);
    poolsum<10><<<CB, 128, 0, stream>>>(sE, entityV, NV, 400, news, NN, 400);
    vert_assemble<<<CB, 224, 0, stream>>>(vert_ids, vert_table, news, NN, b0);

    // final fc3 (WMMA, fp32 output straight into d_out)
    gemm_bf16<true, false><<<dim3((ON + 63) / 64, CB / 64), 256, 0, stream>>>(
        news, NN, fc3Wb, ON, fc3_b, (void*)(out + (size_t)b0 * ON), ON, CB, ON, NN);
  }
}
